// EdgeCell_31628139168216
// MI455X (gfx1250) — matrix operand order repaired and checked by reading, not compiled, not run on hardware
//
#include <hip/hip_runtime.h>
#include <hip/hip_bf16.h>

// ---- sizes (fixed by the reference) ----
#define N_NODES 10000
#define DEG     16
#define E_EDGES (N_NODES * DEG)   // 160000
#define D_IN    128
#define HID     128
#define G4      (4 * HID)         // 512
#define NODES_PER_BLK 4           // 64 edge rows per block

typedef __attribute__((ext_vector_type(16))) __bf16       v16bf;
typedef __attribute__((ext_vector_type(8)))  float        v8f;
typedef __attribute__((ext_vector_type(4)))  float        v4f;
typedef __attribute__((ext_vector_type(4)))  unsigned int v4u;
typedef __attribute__((ext_vector_type(2)))  unsigned int v2u;

// 32-byte WMMA operand fragment viewed as two 16-byte chunks
union Frag {
    v16bf v;
    v4u   q[2];
};

// 4 packed bf16 <-> 8 bytes (for wide LDS stores during staging)
union Pack4 {
    __bf16 b[4];
    v2u    u;
};

// fast activations: one v_exp_f32 TRANS op each, correct saturation
__device__ __forceinline__ float sigf(float x) {
    return 1.0f / (1.0f + __expf(-x));
}
__device__ __forceinline__ float tanh_fast(float x) {
    // 1 - 2/(e^{2x}+1):  x->+inf => 1, x->-inf => -1, no NaN path
    float t = __expf(2.0f * x);
    return 1.0f - 2.0f / (t + 1.0f);
}

// ---------------------------------------------------------------------------
// Kernel 0: W[k][c] (row-major [128 x 512]) -> column-major bf16 hi/lo
//           WT_hi[c][k], WT_lo[c][k]  (bf16x3 error-compensated split)
// ---------------------------------------------------------------------------
__global__ __launch_bounds__(256) void convert_wT_kernel(
    const float* __restrict__ w,        // [D_IN][G4]
    __bf16* __restrict__ wt_hi,         // [G4][D_IN]
    __bf16* __restrict__ wt_lo)         // [G4][D_IN]
{
    int idx = blockIdx.x * 256 + threadIdx.x;      // 0 .. 65535
    if (idx >= D_IN * G4) return;
    int c = idx >> 7;          // 0..511
    int k = idx & (D_IN - 1);  // 0..127
    float  x = w[k * G4 + c];
    __bf16 h = (__bf16)x;
    float  r = x - (float)h;
    wt_hi[c * D_IN + k] = h;
    wt_lo[c * D_IN + k] = (__bf16)r;
}

// ---------------------------------------------------------------------------
// Stage ROWS x 128 fp32 rows into LDS as bf16 hi/lo, vectorized:
// float4 global loads, packed 4xbf16 (b64) LDS stores.
// ---------------------------------------------------------------------------
template<int ROWS>
__device__ __forceinline__ void stage_a(
    const float* __restrict__ src,   // ROWS*128 floats, 16B aligned
    __bf16* a_hi, __bf16* a_lo, int tid)
{
    const v4f* s4 = (const v4f*)src;
    v2u* h2 = (v2u*)a_hi;
    v2u* l2 = (v2u*)a_lo;
    const int nvec = ROWS * D_IN / 4;
    for (int idx = tid; idx < nvec; idx += 256) {
        v4f x = s4[idx];
        Pack4 ph, pl;
#pragma unroll
        for (int j = 0; j < 4; ++j) {
            __bf16 hb = (__bf16)x[j];
            ph.b[j] = hb;
            pl.b[j] = (__bf16)(x[j] - (float)hb);
        }
        h2[idx] = ph.u;
        l2[idx] = pl.u;
    }
}

// ---------------------------------------------------------------------------
// GEMM core: MSUB M-subtiles of 16 rows, 4 N-tiles per wave, K = 128.
// A staged in LDS as bf16 hi/lo rows [MSUB*16][128]; B from L2-resident
// bf16 hi/lo WT.  B fragments are reused across all MSUB subtiles.
// acc = Ah*Bh + Ah*Bl + Al*Bh  (fp32 accumulate; bf16x3 split)
// ---------------------------------------------------------------------------
template<int MSUB>
__device__ __forceinline__ void gemm_core(
    const __bf16* a_hi, const __bf16* a_lo,
    const __bf16* __restrict__ wt_hi,                // [512][128]
    const __bf16* __restrict__ wt_lo,
    const int colbase[4],                            // N base per t-tile
    int lane, v8f acc[MSUB][4])
{
    const int lhalf = (lane >> 4) & 1;   // 0: lanes 0-15, 1: lanes 16-31
    const int lmod  = lane & 15;

    const v4u* ah = (const v4u*)a_hi;
    const v4u* al = (const v4u*)a_lo;
    const v4u* bh = (const v4u*)wt_hi;
    const v4u* bl = (const v4u*)wt_lo;

#pragma unroll 1
    for (int ks = 0; ks < D_IN; ks += 32) {
        // A fragments (16x32 bf16): lanes<16 -> K {0..7}+{16..23},
        //                           lanes>=16 -> K {8..15}+{24..31}
        Frag Ah[MSUB], Al[MSUB];
#pragma unroll
        for (int m = 0; m < MSUB; ++m) {
            const int abase = ((m * 16 + lmod) * D_IN + ks + 8 * lhalf) >> 3;
            Ah[m].q[0] = ah[abase];  Ah[m].q[1] = ah[abase + 2];
            Al[m].q[0] = al[abase];  Al[m].q[1] = al[abase + 2];
        }

#pragma unroll
        for (int t = 0; t < 4; ++t) {
            // B fragment (32x16 bf16): lanes<16 -> K {0..7}+{16..23}, lanes>=16 -> K {8..15}+{24..31}
            const int ncol  = colbase[t] + lmod;
            const int bbase = (ncol * D_IN + ks + 8 * lhalf) >> 3;
            Frag Bh, Bl;
            Bh.q[0] = bh[bbase]; Bh.q[1] = bh[bbase + 2];
            Bl.q[0] = bl[bbase]; Bl.q[1] = bl[bbase + 2];

#pragma unroll
            for (int m = 0; m < MSUB; ++m) {
                acc[m][t] = __builtin_amdgcn_wmma_f32_16x16x32_bf16(
                    false, Ah[m].v, false, Bh.v, (short)0, acc[m][t], false, false);
                acc[m][t] = __builtin_amdgcn_wmma_f32_16x16x32_bf16(
                    false, Ah[m].v, false, Bl.v, (short)0, acc[m][t], false, false);
                acc[m][t] = __builtin_amdgcn_wmma_f32_16x16x32_bf16(
                    false, Al[m].v, false, Bh.v, (short)0, acc[m][t], false, false);
            }
        }
    }
}

// ---------------------------------------------------------------------------
// Kernel 1: node_hh[n][c] = node_h[n] @ W_hh + bias   (fp32 out, to workspace)
// grid = 625 blocks (10000 = 625 * 16 rows), 256 threads
// ---------------------------------------------------------------------------
__global__ __launch_bounds__(256) void node_hh_kernel(
    const float* __restrict__ node_h,    // [N_NODES][HID]
    const __bf16* __restrict__ wt_hi,    // [G4][D_IN]
    const __bf16* __restrict__ wt_lo,
    const float* __restrict__ bias,      // [G4]
    float* __restrict__ node_hh)         // [N_NODES][G4]
{
    __shared__ __bf16 a_hi[16 * D_IN];
    __shared__ __bf16 a_lo[16 * D_IN];

    const int tid   = threadIdx.x;
    const int node0 = blockIdx.x * 16;

    stage_a<16>(node_h + (size_t)node0 * D_IN, a_hi, a_lo, tid);
    __syncthreads();

    const int lane = tid & 31, wave = tid >> 5;
    const int lhalf = (lane >> 4) & 1, lmod = lane & 15;

    int colbase[4];
#pragma unroll
    for (int t = 0; t < 4; ++t) colbase[t] = wave * 64 + t * 16;

    v8f acc[1][4] = {};
    gemm_core<1>(a_hi, a_lo, wt_hi, wt_lo, colbase, lane, acc);

    // C/D layout: vgpr r, lanes<16 -> M=r, lanes>=16 -> M=r+8; N = lane%16.
#pragma unroll
    for (int t = 0; t < 4; ++t) {
        const int col = colbase[t] + lmod;
        const float b = bias[col];
#pragma unroll
        for (int r = 0; r < 8; ++r) {
            const int row = r + 8 * lhalf;
            node_hh[(size_t)(node0 + row) * G4 + col] = acc[0][t][r] + b;
        }
    }
}

// ---------------------------------------------------------------------------
// Kernel 2: block b handles nodes [4b, 4b+4) = 64 edges.
// Wave w's 4 N-tiles map to col = t*128 + 16w, so tile t holds gate t
// (i,f,g,o) for the SAME hidden columns -> register-local LSTM epilogue,
// no gates LDS, no second barrier.
// grid = 2500 blocks, 256 threads
// ---------------------------------------------------------------------------
__global__ __launch_bounds__(256) void edge_cell_kernel(
    const float* __restrict__ edge_i,    // [E][D_IN]
    const __bf16* __restrict__ wt_hi,    // [G4][D_IN]
    const __bf16* __restrict__ wt_lo,
    const float* __restrict__ node_hh,   // [N_NODES][G4]
    const float* __restrict__ node_c,    // [N_NODES][HID]
    float* __restrict__ out_h,           // [E][HID]
    float* __restrict__ out_c)           // [E][HID]
{
    __shared__ __bf16 a_hi[NODES_PER_BLK * 16 * D_IN];   // 16 KB
    __shared__ __bf16 a_lo[NODES_PER_BLK * 16 * D_IN];   // 16 KB

    const int tid   = threadIdx.x;
    const int node0 = blockIdx.x * NODES_PER_BLK;
    const int e0    = node0 * DEG;

    stage_a<NODES_PER_BLK * 16>(edge_i + (size_t)e0 * D_IN, a_hi, a_lo, tid);
    __syncthreads();

    const int lane = tid & 31, wave = tid >> 5;
    const int lhalf = (lane >> 4) & 1, lmod = lane & 15;

    int colbase[4];
#pragma unroll
    for (int t = 0; t < 4; ++t) colbase[t] = t * HID + wave * 16;  // gate t

    v8f acc[NODES_PER_BLK][4] = {};
    gemm_core<NODES_PER_BLK>(a_hi, a_lo, wt_hi, wt_lo, colbase, lane, acc);

    // register-local LSTM epilogue: per node m, per accumulator row
    const int hcol = wave * 16 + lmod;   // hidden column owned by this lane
#pragma unroll
    for (int m = 0; m < NODES_PER_BLK; ++m) {
        const int n = node0 + m;
        const float nh_i = node_hh[(size_t)n * G4 + 0 * HID + hcol];
        const float nh_f = node_hh[(size_t)n * G4 + 1 * HID + hcol];
        const float nh_g = node_hh[(size_t)n * G4 + 2 * HID + hcol];
        const float nh_o = node_hh[(size_t)n * G4 + 3 * HID + hcol];
        const float c0   = node_c[(size_t)n * HID + hcol];

#pragma unroll
        for (int r = 0; r < 8; ++r) {
            const int row = r + 8 * lhalf;            // 0..15 within node m
            const float gi = acc[m][0][r] + nh_i;
            const float gf = acc[m][1][r] + nh_f;
            const float gg = acc[m][2][r] + nh_g;
            const float go = acc[m][3][r] + nh_o;

            const float cc = sigf(gf) * c0 + sigf(gi) * tanh_fast(gg);
            const float hh = sigf(go) * tanh_fast(go); // faithful: o used twice

            const size_t eidx = (size_t)(e0 + m * 16 + row) * HID + hcol;
            out_h[eidx] = hh;
            out_c[eidx] = cc;
        }
    }
}

// ---------------------------------------------------------------------------
// launcher
// ---------------------------------------------------------------------------
extern "C" void kernel_launch(void* const* d_in, const int* in_sizes, int n_in,
                              void* d_out, int out_size, void* d_ws, size_t ws_size,
                              hipStream_t stream)
{
    const float* node_h = (const float*)d_in[0];   // [10000,128]
    const float* node_c = (const float*)d_in[1];   // [10000,128]
    const float* edge_i = (const float*)d_in[2];   // [160000,128]
    const float* w_ih   = (const float*)d_in[3];   // [128,512]
    const float* w_hh   = (const float*)d_in[4];   // [128,512]
    const float* bias   = (const float*)d_in[5];   // [512]
    // d_in[6] segment_ids: implicit (e / 16), not needed

    // workspace: 4 x 64K bf16 weight copies (512 KB) + node_hh fp32 (20.5 MB)
    const size_t WEL = (size_t)D_IN * G4;          // 65536 elements per matrix
    __bf16* wtih_hi = (__bf16*)d_ws;
    __bf16* wtih_lo = wtih_hi + WEL;
    __bf16* wthh_hi = wtih_lo + WEL;
    __bf16* wthh_lo = wthh_hi + WEL;
    float*  node_hh = (float*)(wthh_lo + WEL);     // [10000,512]

    float* out_h = (float*)d_out;                          // [E,128]
    float* out_c = out_h + (size_t)E_EDGES * HID;          // [E,128]

    convert_wT_kernel<<<(D_IN * G4 + 255) / 256, 256, 0, stream>>>(w_ih, wtih_hi, wtih_lo);
    convert_wT_kernel<<<(D_IN * G4 + 255) / 256, 256, 0, stream>>>(w_hh, wthh_hi, wthh_lo);
    node_hh_kernel<<<N_NODES / 16, 256, 0, stream>>>(node_h, wthh_hi, wthh_lo, bias, node_hh);
    edge_cell_kernel<<<N_NODES / NODES_PER_BLK, 256, 0, stream>>>(
        edge_i, wtih_hi, wtih_lo, node_hh, node_c, out_h, out_c);
}
